// LSTM_Decoder_19971597927086
// MI455X (gfx1250) — compile-verified, matrix operand order checked by reading, not run on hardware
//
#include <hip/hip_runtime.h>
#include <hip/hip_bf16.h>

#define BATCH 128
#define SEQ   512
#define IDIM  768
#define HDIM  768
#define GDIM  3072           // 4*H
#define KTILES 24            // 768 / 32
#define NTILES 192           // 3072 / 16
#define PACK_ELEMS (NTILES * KTILES * 32 * 16)   // 2,359,296 bf16 per weight matrix

typedef __attribute__((ext_vector_type(16))) __bf16 v16bf;
typedef __attribute__((ext_vector_type(8)))  __bf16 v8bf;
typedef __attribute__((ext_vector_type(4)))  __bf16 v4bf;
typedef __attribute__((ext_vector_type(8)))  float  v8f;

__device__ __forceinline__ v8f wmma_bf16(v16bf a, v16bf b, v8f c) {
  // D = A(16x32 bf16) * B(32x16 bf16) + C(16x16 f32)
  return __builtin_amdgcn_wmma_f32_16x16x32_bf16(false, a, false, b, (short)0, c,
                                                 false, false);
}
__device__ __forceinline__ float sigf(float x) {
  return 1.0f / (1.0f + __expf(-x));
}
__device__ __forceinline__ float tanhfast(float x) {
  x = fminf(fmaxf(x, -15.0f), 15.0f);
  float e = __expf(-2.0f * x);
  return (1.0f - e) / (1.0f + e);
}

// ---------------------------------------------------------------------------
// One-time: pack W[3072,768] f32 row-major into bf16 WMMA B-fragment layout.
// Fragment (nt, kt): 32 lanes x 16 bf16.  Lane l holds column n = nt*16+(l&15);
// elems 0..7 -> K = kt*32 + (l>>4)*8 + e ; elems 8..15 -> +16.
// ---------------------------------------------------------------------------
__global__ __launch_bounds__(256)
void pack_w(const float* __restrict__ W, __bf16* __restrict__ P) {
  int f = blockIdx.x * 256 + threadIdx.x;          // fragment-lane id
  if (f >= NTILES * KTILES * 32) return;
  int lane = f & 31;
  int kt   = (f >> 5) % KTILES;
  int nt   = (f >> 5) / KTILES;
  int grp  = (lane >> 4) * 8;
  int n    = nt * 16 + (lane & 15);
  const float* wr = W + (size_t)n * HDIM + kt * 32;
  v16bf o;
#pragma unroll
  for (int e = 0; e < 8; ++e) o[e] = (__bf16)wr[grp + e];
#pragma unroll
  for (int e = 0; e < 8; ++e) o[8 + e] = (__bf16)wr[16 + grp + e];
  *(v16bf*)(P + (size_t)f * 16) = o;
}

// ---------------------------------------------------------------------------
// One-time: fold biases, init h/c state (into h buffer index 0).
// ---------------------------------------------------------------------------
__global__ __launch_bounds__(256)
void prep_state(const float* __restrict__ h_init, const float* __restrict__ c_init,
                const float* __restrict__ bih0, const float* __restrict__ bhh0,
                const float* __restrict__ bih1, const float* __restrict__ bhh1,
                __bf16* __restrict__ h0b, __bf16* __restrict__ h1b,
                float* __restrict__ c0, float* __restrict__ c1,
                float* __restrict__ bias0, float* __restrict__ bias1) {
  int idx = blockIdx.x * 256 + threadIdx.x;
  if (idx < BATCH * HDIM) {
    float h = h_init[idx], c = c_init[idx];
    h0b[idx] = (__bf16)h;
    h1b[idx] = (__bf16)h;
    c0[idx] = c;
    c1[idx] = c;
  }
  if (idx < GDIM) {
    bias0[idx] = bih0[idx] + bhh0[idx];
    bias1[idx] = bih1[idx] + bhh1[idx];
  }
}

// ---------------------------------------------------------------------------
// Fully fused LSTM layer step:
//   gates = A0 @ B0pack + A1 @ B1pack + bias   (dual K-accumulated GEMM, WMMA)
//   then cell update (i,f,g,o) entirely in LDS — gates never touch memory.
//
// Tiling: WG (bx, by) owns batch rows [by*32, by*32+32) x hidden units
// [bx*32, bx*32+32).  For that it computes gate columns {n, n+H, n+2H, n+3H}:
// wave pair q (wq = wave>>1) computes gate q's two 16-col tiles
// (nt0 = q*48 + bx*2), wm = wave&1 picks the 16-row M tile.
// LDS: A-stage buffer (32x768 bf16, 49.7KB) aliased with the 4x32x33 f32
// gate tile (16.9KB) used by the fused cell epilogue.
// ---------------------------------------------------------------------------
__global__ __launch_bounds__(256)
void lstm_layer(const float* __restrict__ af0, const __bf16* __restrict__ ab0,
                long long as0,
                const __bf16* __restrict__ ab1,
                const __bf16* __restrict__ B0, const __bf16* __restrict__ B1,
                const float* __restrict__ bias,
                float* __restrict__ c, __bf16* __restrict__ hb_out,
                float* __restrict__ hout, long long hstride) {
  __shared__ __align__(16) unsigned char smem[32 * (HDIM + 8) * sizeof(__bf16)];
  auto As = (__bf16(*)[HDIM + 8])smem;             // A-stage: 32 x 776 bf16
  auto Gs = (float(*)[32][33])smem;                // gate tile: [4][32][33] f32

  const int tid  = threadIdx.x;
  const int lane = tid & 31;
  const int wave = tid >> 5;
  const int wm   = wave & 1;                       // M sub-tile
  const int wq   = wave >> 1;                      // gate index 0..3 (i,f,g,o)
  const int rowBase = blockIdx.y * 32;             // batch-row base
  const int nt0  = wq * 48 + blockIdx.x * 2;       // gate q, hidden block bx
  const int laneN = lane & 15;
  const int kgrp  = (lane >> 4) * 8;               // K-lane-group offset
  const int rowA  = wm * 16 + laneN;               // row in LDS A tile

  v8f acc0 = {};
  v8f acc1 = {};

  for (int g = 0; g < 2; ++g) {
    if (g) __syncthreads();
    // ---- stage A tile (32 x 768) into LDS as bf16 ----
    if (g == 0 && af0) {
      for (int cc = tid; cc < 32 * (HDIM / 4); cc += 256) {
        int r  = cc / (HDIM / 4);
        int kk = (cc % (HDIM / 4)) * 4;
        float4 v = *(const float4*)(af0 + (size_t)(rowBase + r) * (size_t)as0 + kk);
        v4bf w;
        w[0] = (__bf16)v.x; w[1] = (__bf16)v.y;
        w[2] = (__bf16)v.z; w[3] = (__bf16)v.w;
        *(v4bf*)&As[r][kk] = w;
      }
    } else {
      const __bf16* src = g ? ab1 : ab0;
      for (int cc = tid; cc < 32 * (HDIM / 8); cc += 256) {
        int r  = cc / (HDIM / 8);
        int kk = (cc % (HDIM / 8)) * 8;
        *(v8bf*)&As[r][kk] =
            *(const v8bf*)(src + (size_t)(rowBase + r) * (size_t)HDIM + kk);
      }
    }
    __syncthreads();

    const __bf16* bp0 = (g ? B1 : B0) + (size_t)nt0 * KTILES * 512 + lane * 16;
    const __bf16* bp1 = bp0 + (size_t)KTILES * 512;

    for (int kt = 0; kt < KTILES; ++kt) {
      const int kk = kt * 32;
      v16bf a, b0, b1;
      ((v8bf*)&a)[0]  = *(const v8bf*)&As[rowA][kk + kgrp];
      ((v8bf*)&a)[1]  = *(const v8bf*)&As[rowA][kk + 16 + kgrp];
      ((v8bf*)&b0)[0] = *(const v8bf*)(bp0 + (size_t)kt * 512);
      ((v8bf*)&b0)[1] = *(const v8bf*)(bp0 + (size_t)kt * 512 + 8);
      ((v8bf*)&b1)[0] = *(const v8bf*)(bp1 + (size_t)kt * 512);
      ((v8bf*)&b1)[1] = *(const v8bf*)(bp1 + (size_t)kt * 512 + 8);
      __builtin_prefetch(bp0 + (size_t)(kt + 1) * 512, 0, 1);
      __builtin_prefetch(bp1 + (size_t)(kt + 1) * 512, 0, 1);
      acc0 = wmma_bf16(a, b0, acc0);
      acc1 = wmma_bf16(a, b1, acc1);
    }
  }

  // ---- drop gate tile into LDS (C layout: VGPR r -> M=base+r / base+8+r) ----
  __syncthreads();                                  // done reading As
  const int   gr  = wm * 16 + ((lane >> 4) ? 8 : 0);
  const float bv0 = bias[nt0 * 16 + laneN];
  const float bv1 = bias[(nt0 + 1) * 16 + laneN];
#pragma unroll
  for (int r = 0; r < 8; ++r) {
    Gs[wq][gr + r][laneN]      = acc0[r] + bv0;
    Gs[wq][gr + r][16 + laneN] = acc1[r] + bv1;
  }
  __syncthreads();

  // ---- fused LSTM cell over the 32x32 (batch x hidden) tile ----
  for (int e = tid; e < 32 * 32; e += 256) {
    int r  = e >> 5;
    int nl = e & 31;
    int b  = rowBase + r;
    int n  = blockIdx.x * 32 + nl;
    float ig = sigf(Gs[0][r][nl]);
    float fg = sigf(Gs[1][r][nl]);
    float gg = tanhfast(Gs[2][r][nl]);
    float og = sigf(Gs[3][r][nl]);
    size_t ci = (size_t)b * HDIM + n;
    float cn = fg * c[ci] + ig * gg;
    c[ci] = cn;
    float h = og * tanhfast(cn);
    hb_out[ci] = (__bf16)h;
    if (hout) hout[(size_t)b * (size_t)hstride + n] = h;
  }
}

// Append final (h1, c1) after out[B,S,H] in d_out.
__global__ __launch_bounds__(256)
void final_copy(const float* __restrict__ out, const float* __restrict__ c1,
                float* __restrict__ tail) {
  int idx = blockIdx.x * 256 + threadIdx.x;
  if (idx >= BATCH * HDIM) return;
  int b = idx / HDIM, n = idx - b * HDIM;
  tail[idx] = out[(size_t)b * SEQ * HDIM + (size_t)(SEQ - 1) * HDIM + n];
  tail[BATCH * HDIM + idx] = c1[idx];
}

// ---------------------------------------------------------------------------
extern "C" void kernel_launch(void* const* d_in, const int* in_sizes, int n_in,
                              void* d_out, int out_size, void* d_ws, size_t ws_size,
                              hipStream_t stream) {
  (void)in_sizes; (void)n_in; (void)out_size; (void)ws_size;
  const float* x      = (const float*)d_in[0];
  const float* h_init = (const float*)d_in[1];
  const float* c_init = (const float*)d_in[2];
  const float* W_ih0  = (const float*)d_in[3];
  const float* W_hh0  = (const float*)d_in[4];
  const float* b_ih0  = (const float*)d_in[5];
  const float* b_hh0  = (const float*)d_in[6];
  const float* W_ih1  = (const float*)d_in[7];
  const float* W_hh1  = (const float*)d_in[8];
  const float* b_ih1  = (const float*)d_in[9];
  const float* b_hh1  = (const float*)d_in[10];
  float* outp = (float*)d_out;

  char*  ws  = (char*)d_ws;
  size_t off = 0;
  auto carve = [&](size_t bytes) -> void* {
    void* p = ws + off;
    off = (off + bytes + 255) & ~(size_t)255;
    return p;
  };
  const size_t packB = (size_t)PACK_ELEMS * sizeof(__bf16);
  __bf16* Wih0p = (__bf16*)carve(packB);
  __bf16* Whh0p = (__bf16*)carve(packB);
  __bf16* Wih1p = (__bf16*)carve(packB);
  __bf16* Whh1p = (__bf16*)carve(packB);
  float*  bias0 = (float*)carve(GDIM * sizeof(float));
  float*  bias1 = (float*)carve(GDIM * sizeof(float));
  float*  c0 = (float*)carve((size_t)BATCH * HDIM * sizeof(float));
  float*  c1 = (float*)carve((size_t)BATCH * HDIM * sizeof(float));
  __bf16* h0b[2], *h1b[2];
  h0b[0] = (__bf16*)carve((size_t)BATCH * HDIM * sizeof(__bf16));
  h0b[1] = (__bf16*)carve((size_t)BATCH * HDIM * sizeof(__bf16));
  h1b[0] = (__bf16*)carve((size_t)BATCH * HDIM * sizeof(__bf16));
  h1b[1] = (__bf16*)carve((size_t)BATCH * HDIM * sizeof(__bf16));

  const int packBlocks = (NTILES * KTILES * 32 + 255) / 256;   // 576
  pack_w<<<packBlocks, 256, 0, stream>>>(W_ih0, Wih0p);
  pack_w<<<packBlocks, 256, 0, stream>>>(W_hh0, Whh0p);
  pack_w<<<packBlocks, 256, 0, stream>>>(W_ih1, Wih1p);
  pack_w<<<packBlocks, 256, 0, stream>>>(W_hh1, Whh1p);

  const int ewBlocks = (BATCH * HDIM + 255) / 256;             // 384
  prep_state<<<ewBlocks, 256, 0, stream>>>(h_init, c_init, b_ih0, b_hh0,
                                           b_ih1, b_hh1, h0b[0], h1b[0],
                                           c0, c1, bias0, bias1);

  const dim3 grid(HDIM / 32, BATCH / 32);                      // (24, 4)
  for (int t = 0; t < SEQ; ++t) {
    const int pi = t & 1;        // h read buffer
    const int ni = pi ^ 1;       // h write buffer
    // layer 0: gates = x_t @ Wih0^T + h0_old @ Whh0^T + bias0 ; cell -> h0_new
    lstm_layer<<<grid, 256, 0, stream>>>(
        x + (size_t)t * IDIM, nullptr, (long long)SEQ * IDIM,
        h0b[pi], Wih0p, Whh0p, bias0, c0, h0b[ni], nullptr, 0);
    // layer 1: gates = h0_new @ Wih1^T + h1_old @ Whh1^T + bias1 ; cell -> h1_new
    lstm_layer<<<grid, 256, 0, stream>>>(
        nullptr, h0b[ni], (long long)HDIM,
        h1b[pi], Wih1p, Whh1p, bias1, c1, h1b[ni],
        outp + (size_t)t * HDIM, (long long)SEQ * HDIM);
  }
  final_copy<<<ewBlocks, 256, 0, stream>>>(outp, c1,
                                           outp + (size_t)BATCH * SEQ * HDIM);
}
